// CNN_2_EDropout_78134045048896
// MI455X (gfx1250) — matrix operand order repaired and checked by reading, not compiled, not run on hardware
//
#include <hip/hip_runtime.h>
#include <stdint.h>

// ---------------------------------------------------------------------------
// Types for CDNA5 WMMA (gfx1250, wave32)
// ---------------------------------------------------------------------------
typedef __attribute__((ext_vector_type(16))) __bf16        v16bf;
typedef __attribute__((ext_vector_type(8)))  float         v8f;
typedef __attribute__((ext_vector_type(8)))  unsigned int  v8u;

#define EP_RELU    1
#define EP_BIAS_N  2
#define EP_BIAS_M  4
#define EP_STORE_T 8   // store bf16 output transposed: outb[n*ldob + m] (NHWC conv output)

__device__ __forceinline__ unsigned short f2bf(float f) {
  union { float f; unsigned int u; } c; c.f = f;
  unsigned int u = c.u;
  u += 0x7FFFu + ((u >> 16) & 1u);   // round-to-nearest-even
  return (unsigned short)(u >> 16);
}
__device__ __forceinline__ float bf2f(unsigned short h) {
  union { unsigned int u; float f; } c; c.u = ((unsigned int)h) << 16;
  return c.f;
}

// ---------------------------------------------------------------------------
// WMMA fragment loaders (contiguous k order k = 16*half + s, same for A and B)
// A (16x32 bf16, row-major, K-contiguous): lane m = lane&15, half = lane>>4
//   half=0: V0..7 -> K=0..15 ; half=1: K=16..31  => contiguous loads (same k order as B).
// B (32x16 bf16 from Bt[N][K] row-major): lane n = lane&15
//   half=0: V0..7 -> K=0..15 ; half=1: K=16..31  => two contiguous 16B loads.
// All operand buffers are padded so no load guards are required.
// ---------------------------------------------------------------------------
__device__ __forceinline__ v16bf loadA(const unsigned short* A, int k0, int half) {
  v8u u;
#pragma unroll
  for (int v = 0; v < 8; ++v) {
    int kb = 2 * v + 16 * half;
    u[v] = *(const unsigned int*)(A + k0 + kb);
  }
  return __builtin_bit_cast(v16bf, u);
}

__device__ __forceinline__ v16bf loadB(const unsigned short* Bp, int k0, int half) {
  v8u u;
#pragma unroll
  for (int v = 0; v < 8; ++v) {
    int kb = 2 * v + 16 * half;
    u[v] = *(const unsigned int*)(Bp + k0 + kb);
  }
  return __builtin_bit_cast(v16bf, u);
}

// ---------------------------------------------------------------------------
// Generic bf16 WMMA GEMM: C[M,N] = A[M,K] * Bt[N,K]^T  (fp32 accumulate)
// One wave computes a 16(M) x 64(N) strip: 4 accumulators, A-fragment reuse x4.
// Software-pipelined: step t+1 fragments are loaded before step t's WMMAs so
// the loadcnt wait in front of the WMMAs covers loads issued a full
// iteration earlier (latency overlapped with matrix work).
// Requires: M % 16 == 0, K % 32 == 0, Bt has at least ceil(N/64)*64 rows
// (zero-padded).
// ---------------------------------------------------------------------------
__global__ void gemm_bf16_kernel(const unsigned short* __restrict__ A, int lda,
                                 const unsigned short* __restrict__ Bt, int ldb,
                                 float* __restrict__ outf, int ldof,
                                 unsigned short* __restrict__ outb, int ldob,
                                 const float* __restrict__ bias,
                                 const float* __restrict__ mul, int ldmul,
                                 int M, long long N, int K, int flags) {
  int lane = threadIdx.x & 31;
  int wave = blockIdx.x * (blockDim.x >> 5) + (threadIdx.x >> 5);
  int Mt = M >> 4;
  long long Nt = (N + 63) >> 6;
  if (wave >= (long long)Mt * Nt) return;           // wave-uniform: EXEC all-1s below
  int      tm = wave % Mt;
  long long tn = wave / Mt;
  int half = lane >> 4, l15 = lane & 15;
  const unsigned short* Ap  = A + (size_t)(tm * 16 + l15) * lda;
  const unsigned short* Bp0 = Bt + (size_t)(tn * 64 + l15) * ldb;
  const unsigned short* Bp1 = Bp0 + (size_t)16 * ldb;
  const unsigned short* Bp2 = Bp0 + (size_t)32 * ldb;
  const unsigned short* Bp3 = Bp0 + (size_t)48 * ldb;

  v8f acc[4] = {};
  v16bf a  = loadA(Ap, 0, half);
  v16bf b0 = loadB(Bp0, 0, half);
  v16bf b1 = loadB(Bp1, 0, half);
  v16bf b2 = loadB(Bp2, 0, half);
  v16bf b3 = loadB(Bp3, 0, half);
  for (int k0 = 32; k0 < K; k0 += 32) {
    v16bf an  = loadA(Ap, k0, half);
    v16bf bn0 = loadB(Bp0, k0, half);
    v16bf bn1 = loadB(Bp1, k0, half);
    v16bf bn2 = loadB(Bp2, k0, half);
    v16bf bn3 = loadB(Bp3, k0, half);
    acc[0] = __builtin_amdgcn_wmma_f32_16x16x32_bf16(false, a, false, b0, (short)0, acc[0], false, false);
    acc[1] = __builtin_amdgcn_wmma_f32_16x16x32_bf16(false, a, false, b1, (short)0, acc[1], false, false);
    acc[2] = __builtin_amdgcn_wmma_f32_16x16x32_bf16(false, a, false, b2, (short)0, acc[2], false, false);
    acc[3] = __builtin_amdgcn_wmma_f32_16x16x32_bf16(false, a, false, b3, (short)0, acc[3], false, false);
    a = an; b0 = bn0; b1 = bn1; b2 = bn2; b3 = bn3;
  }
  acc[0] = __builtin_amdgcn_wmma_f32_16x16x32_bf16(false, a, false, b0, (short)0, acc[0], false, false);
  acc[1] = __builtin_amdgcn_wmma_f32_16x16x32_bf16(false, a, false, b1, (short)0, acc[1], false, false);
  acc[2] = __builtin_amdgcn_wmma_f32_16x16x32_bf16(false, a, false, b2, (short)0, acc[2], false, false);
  acc[3] = __builtin_amdgcn_wmma_f32_16x16x32_bf16(false, a, false, b3, (short)0, acc[3], false, false);

#pragma unroll
  for (int f = 0; f < 4; ++f) {
    long long n = tn * 64 + f * 16 + l15;
    if (n < N) {
#pragma unroll
      for (int r = 0; r < 8; ++r) {
        int m = tm * 16 + r + 8 * half;        // C layout: VGPR r, lanes16-31 => M+8
        float v = acc[f][r];
        if (flags & EP_BIAS_N) v += bias[n];
        if (flags & EP_BIAS_M) v += bias[m];
        if (flags & EP_RELU)   v = fmaxf(v, 0.0f);
        if (mul) v *= mul[(size_t)m * ldmul + n];
        if (outf) outf[(size_t)m * ldof + n] = v;
        if (outb) {
          if (flags & EP_STORE_T) outb[(size_t)n * ldob + m] = f2bf(v);
          else                    outb[(size_t)m * ldob + n] = f2bf(v);
        }
      }
    }
  }
}

// ---------------------------------------------------------------------------
// Implicit-GEMM 5x5 conv, pad=2, stride 1, NHWC bf16 in/out, Cin % 32 == 0.
// Weights pre-reordered: Wr[kyx][Cout][Cin]. One wave: 16(Cout) x 64(pixels).
// The 25-tap x (Cin/32)-step loop is flattened and software-pipelined.
// Per-fragment tap validity lives in a 25-bit mask (built once from separable
// row/col patterns); the per-tap input offset is lane-uniform SALU work.
// Boundary loads are branchless: clamped safe base + cndmask zero-fill.
// ---------------------------------------------------------------------------
__global__ void conv5x5_wmma_kernel(const unsigned short* __restrict__ in,
                                    const unsigned short* __restrict__ Wr,
                                    const float* __restrict__ bias,
                                    unsigned short* __restrict__ out,
                                    int B_, int H, int Wd, int Cin, int Cout) {
  int lane = threadIdx.x & 31;
  int wave = blockIdx.x * (blockDim.x >> 5) + (threadIdx.x >> 5);
  int Mt = Cout >> 4;
  long long Npix = (long long)B_ * H * Wd;          // multiple of 64 here
  long long Nt = Npix >> 6;
  if (wave >= (long long)Mt * Nt) return;
  int      tm = wave % Mt;
  long long tn = wave / Mt;
  int half = lane >> 4, l15 = lane & 15;
  int arow = tm * 16 + l15;
  const unsigned short* Abase = Wr + (size_t)arow * Cin;   // + kyx*Cout*Cin
  const size_t Astride = (size_t)Cout * Cin;

  long long npx[4], base0[4];
  unsigned vmask[4];
#pragma unroll
  for (int f = 0; f < 4; ++f) {
    long long nn = tn * 64 + f * 16 + l15;
    npx[f] = nn;
    base0[f] = nn * Cin;
    int xq = (int)(nn % Wd);
    long long t = nn / Wd;
    int yq = (int)(t % H);
    unsigned rowpat = 0, colpat = 0;
#pragma unroll
    for (int i = 0; i < 5; ++i) {
      int yy = yq + i - 2, xx = xq + i - 2;
      rowpat |= (yy >= 0 && yy < H)  ? (1u << i) : 0u;
      colpat |= (xx >= 0 && xx < Wd) ? (1u << i) : 0u;
    }
    unsigned m = 0;
#pragma unroll
    for (int i = 0; i < 5; ++i)
      m |= (rowpat & (1u << i)) ? (colpat << (5 * i)) : 0u;
    vmask[f] = m;                 // bit (ky*5+kx) = tap validity
  }

  const int ksteps = Cin >> 5;
  const int T = 25 * ksteps;

  // ---- load step 0 (kyx = 0 i.e. dy=-2,dx=-2 ; kc = 0) ----
  v16bf a, b[4];
  {
    a = loadA(Abase, 0, half);
    long long soff = ((long long)(-2) * Wd + (-2)) * Cin;
#pragma unroll
    for (int f = 0; f < 4; ++f) {
      bool ok = (vmask[f] & 1u) != 0;
      long long ab = ok ? (base0[f] + soff) : 0;   // safe base; masked below
      v8u u;
#pragma unroll
      for (int v = 0; v < 8; ++v) {
        int kb = 2 * v + 16 * half;
        unsigned int tv = *(const unsigned int*)(in + ab + kb);
        u[v] = ok ? tv : 0u;
      }
      b[f] = __builtin_bit_cast(v16bf, u);
    }
  }

  v8f acc[4] = {};
  // prefetch indices -> step 1
  int pky = 0, pkx = 0, pkc = 0;
  if (++pkc == ksteps) { pkc = 0; if (++pkx == 5) { pkx = 0; ++pky; } }

  for (int t = 0; t < T - 1; ++t) {
    int kyxP = pky * 5 + pkx;
    v16bf an = loadA(Abase + (size_t)kyxP * Astride, pkc * 32, half);
    long long soff = ((long long)(pky - 2) * Wd + (pkx - 2)) * Cin + pkc * 32;
    v16bf bn[4];
#pragma unroll
    for (int f = 0; f < 4; ++f) {
      bool ok = ((vmask[f] >> kyxP) & 1u) != 0;
      long long ab = ok ? (base0[f] + soff) : 0;
      v8u u;
#pragma unroll
      for (int v = 0; v < 8; ++v) {
        int kb = 2 * v + 16 * half;
        unsigned int tv = *(const unsigned int*)(in + ab + kb);
        u[v] = ok ? tv : 0u;
      }
      bn[f] = __builtin_bit_cast(v16bf, u);
    }
    acc[0] = __builtin_amdgcn_wmma_f32_16x16x32_bf16(false, a, false, b[0], (short)0, acc[0], false, false);
    acc[1] = __builtin_amdgcn_wmma_f32_16x16x32_bf16(false, a, false, b[1], (short)0, acc[1], false, false);
    acc[2] = __builtin_amdgcn_wmma_f32_16x16x32_bf16(false, a, false, b[2], (short)0, acc[2], false, false);
    acc[3] = __builtin_amdgcn_wmma_f32_16x16x32_bf16(false, a, false, b[3], (short)0, acc[3], false, false);
    a = an; b[0] = bn[0]; b[1] = bn[1]; b[2] = bn[2]; b[3] = bn[3];
    if (++pkc == ksteps) { pkc = 0; if (++pkx == 5) { pkx = 0; ++pky; } }
  }
  acc[0] = __builtin_amdgcn_wmma_f32_16x16x32_bf16(false, a, false, b[0], (short)0, acc[0], false, false);
  acc[1] = __builtin_amdgcn_wmma_f32_16x16x32_bf16(false, a, false, b[1], (short)0, acc[1], false, false);
  acc[2] = __builtin_amdgcn_wmma_f32_16x16x32_bf16(false, a, false, b[2], (short)0, acc[2], false, false);
  acc[3] = __builtin_amdgcn_wmma_f32_16x16x32_bf16(false, a, false, b[3], (short)0, acc[3], false, false);

  // bias[co] + relu -> NHWC bf16
#pragma unroll
  for (int f = 0; f < 4; ++f) {
#pragma unroll
    for (int r = 0; r < 8; ++r) {
      int m = tm * 16 + r + 8 * half;
      float v = fmaxf(acc[f][r] + bias[m], 0.0f);
      out[(size_t)npx[f] * Cout + m] = f2bf(v);
    }
  }
}

// ---------------------------------------------------------------------------
// Helper / elementwise kernels
// ---------------------------------------------------------------------------
// Pack fp32 -> bf16 with zero padding, optional transpose, optional max(.,0)
__global__ void pack_bf16_kernel(unsigned short* dst, const float* src,
                                 int srcR, int srcC, int dR, int dC,
                                 int transpose, int clamp0) {
  long long idx = (long long)blockIdx.x * blockDim.x + threadIdx.x;
  long long total = (long long)dR * dC;
  if (idx >= total) return;
  int i = (int)(idx / dC), j = (int)(idx % dC);
  float v = 0.0f;
  if (!transpose) { if (i < srcR && j < srcC) v = src[(size_t)i * srcC + j]; }
  else            { if (j < srcR && i < srcC) v = src[(size_t)j * srcC + i]; }
  if (clamp0) v = fmaxf(v, 0.0f);
  dst[idx] = f2bf(v);
}

// OIHW fp32 -> [kyx][Cout][Cin] bf16
__global__ void pack_wr_kernel(unsigned short* dst, const float* w, int Cout, int Cin) {
  long long idx = (long long)blockIdx.x * blockDim.x + threadIdx.x;
  long long total = 25LL * Cout * Cin;
  if (idx >= total) return;
  int kyx = (int)(idx / ((long long)Cout * Cin));
  long long rem = idx % ((long long)Cout * Cin);
  int co = (int)(rem / Cin), ci = (int)(rem % Cin);
  dst[idx] = f2bf(w[((size_t)co * Cin + ci) * 25 + kyx]);
}

// conv1 im2col: x NCHW [512,3,32,32] fp32 -> patches [512*30*30][96] bf16 (K padded 75->96)
__global__ void im2col1_kernel(unsigned short* dst, const float* x) {
  long long idx = (long long)blockIdx.x * blockDim.x + threadIdx.x;
  const long long total = 512LL * 900 * 96;
  if (idx >= total) return;
  int k = (int)(idx % 96);
  long long n = idx / 96;
  int xo = (int)(n % 30);
  long long t = n / 30;
  int yo = (int)(t % 30);
  int b  = (int)(t / 30);
  float v = 0.0f;
  if (k < 75) {
    int ci = k / 25, r = k % 25, ky = r / 5, kx = r % 5;
    int yi = yo + ky - 1, xi = xo + kx - 1;
    if (yi >= 0 && yi < 32 && xi >= 0 && xi < 32)
      v = x[(((size_t)b * 3 + ci) * 32 + yi) * 32 + xi];
  }
  dst[idx] = f2bf(v);
}

// maxpool 3x3 stride 2 VALID, NHWC bf16 -> NHWC bf16
__global__ void maxpool3s2_kernel(const unsigned short* in, unsigned short* out,
                                  int B_, int H, int Wd, int C, int Ho, int Wo) {
  long long idx = (long long)blockIdx.x * blockDim.x + threadIdx.x;
  long long total = (long long)B_ * Ho * Wo * C;
  if (idx >= total) return;
  int c = (int)(idx % C);
  long long t = idx / C;
  int xo = (int)(t % Wo); t /= Wo;
  int yo = (int)(t % Ho);
  int b  = (int)(t / Ho);
  float m = -3.4e38f;
#pragma unroll
  for (int dy = 0; dy < 3; ++dy)
#pragma unroll
    for (int dx = 0; dx < 3; ++dx) {
      int y = yo * 2 + dy, x = xo * 2 + dx;
      m = fmaxf(m, bf2f(in[(((size_t)b * H + y) * Wd + x) * C + c]));
    }
  out[idx] = f2bf(m);
}

// h3 NHWC [512,2,2,256] -> h4 [512,1024] with NCHW flatten order (c*4 + y*2 + x)
__global__ void flatten_kernel(const unsigned short* h3, unsigned short* h4) {
  int idx = blockIdx.x * blockDim.x + threadIdx.x;
  if (idx >= 512 * 1024) return;
  int k = idx & 1023, b = idx >> 10;
  int c = k >> 2, y = (k >> 1) & 1, x = k & 1;
  h4[idx] = h3[(((size_t)b * 2 + y) * 2 + x) * 256 + c];
}

// s = where(z>0, one_hot(label)/z, 0)  -> bf16 [512][32] (K padded 10->32)
__global__ void s1_kernel(const float* z, const int* label, unsigned short* sb) {
  int idx = blockIdx.x * blockDim.x + threadIdx.x;
  if (idx >= 512 * 32) return;
  int j = idx & 31, b = idx >> 5;
  float v = 0.0f;
  if (j < 10 && j == label[b]) {
    float zz = z[b * 16 + j];
    if (zz > 0.0f) v = 1.0f / zz;
  }
  sb[idx] = f2bf(v);
}

// s2 = where(z2>0, p_h6/z2, 0) -> bf16 [512][2048]
__global__ void s2_kernel(const float* z2, const float* ph6, unsigned short* s2b) {
  long long idx = (long long)blockIdx.x * blockDim.x + threadIdx.x;
  if (idx >= 512LL * 2048) return;
  float zz = z2[idx];
  float v = (zz > 0.0f) ? (ph6[idx] / zz) : 0.0f;
  s2b[idx] = f2bf(v);
}

// Excitation dropout: retain_p from pebs, mask with noise, scale h5 -> bf16
__global__ void edrop_kernel(const float* pebs, const float* noise, const float* h5,
                             unsigned short* hed) {
  long long idx = (long long)blockIdx.x * blockDim.x + threadIdx.x;
  if (idx >= 512LL * 2048) return;
  float p = pebs[idx];
  float num = 0.5f * 2047.0f * p;                      // (1-P)*(N-1)*pebs, P=0.5, N=2048
  float retain = 1.0f - num / (num + 0.5f * (1.0f - p));
  float mask = (noise[idx] < retain) ? 1.0f : 0.0f;
  float v = (retain > 0.0f) ? (h5[idx] * mask / retain) : 0.0f;
  hed[idx] = f2bf(v);
}

// ---------------------------------------------------------------------------
// Host orchestration
// ---------------------------------------------------------------------------
extern "C" void kernel_launch(void* const* d_in, const int* in_sizes, int n_in,
                              void* d_out, int out_size, void* d_ws, size_t ws_size,
                              hipStream_t stream) {
  (void)in_sizes; (void)n_in; (void)out_size; (void)ws_size;
  const float* x       = (const float*)d_in[0];
  const float* noise   = (const float*)d_in[1];
  const float* conv1_w = (const float*)d_in[2];
  const float* conv1_b = (const float*)d_in[3];
  const float* conv2_w = (const float*)d_in[4];
  const float* conv2_b = (const float*)d_in[5];
  const float* conv3_w = (const float*)d_in[6];
  const float* conv3_b = (const float*)d_in[7];
  const float* fc1_w   = (const float*)d_in[8];
  const float* fc1_b   = (const float*)d_in[9];
  const float* fc2_w   = (const float*)d_in[10];
  const float* fc2_b   = (const float*)d_in[11];
  const float* fc3_w   = (const float*)d_in[12];
  const float* fc3_b   = (const float*)d_in[13];
  const int*   label   = (const int*)d_in[14];
  float* out = (float*)d_out;

  typedef unsigned short u16;
  char* W8 = (char*)d_ws;
  size_t off = 0;
  auto alloc = [&](size_t bytes) -> char* {
    char* r = W8 + off;
    off += (bytes + 255) & ~(size_t)255;
    return r;
  };
  // persistent buffers
  u16* h1    = (u16*)alloc(512ULL*14*14*96*2);
  u16* h2    = (u16*)alloc(512ULL*6*6*128*2);
  u16* h3    = (u16*)alloc(512ULL*2*2*256*2);
  u16* h4    = (u16*)alloc(512ULL*1024*2);
  u16* w1r   = (u16*)alloc(96ULL*96*2);
  u16* wr2   = (u16*)alloc(25ULL*128*96*2);
  u16* wr3   = (u16*)alloc(25ULL*256*128*2);
  u16* fc1wb = (u16*)alloc(2048ULL*1024*2);
  u16* fc2wb = (u16*)alloc(2048ULL*2048*2);
  u16* fc3wb = (u16*)alloc(64ULL*2048*2);    // 10 real rows, zero-padded to 64
  u16* wp3   = (u16*)alloc(64ULL*2048*2);    // 10 real rows, zero-padded to 64
  u16* wp3t  = (u16*)alloc(2048ULL*32*2);
  u16* wp2   = (u16*)alloc(2048ULL*2048*2);
  u16* wp2t  = (u16*)alloc(2048ULL*2048*2);
  float* h5f   = (float*)alloc(512ULL*2048*4);
  u16*   h5b   = (u16*)  alloc(512ULL*2048*2);
  float* h6ebf = (float*)alloc(512ULL*2048*4);
  u16*   h6ebb = (u16*)  alloc(512ULL*2048*2);
  float* zf    = (float*)alloc(512ULL*16*4);
  u16*   sb    = (u16*)  alloc(512ULL*32*2);
  float* ph6f  = (float*)alloc(512ULL*2048*4);
  float* z2f   = (float*)alloc(512ULL*2048*4);
  u16*   s2b   = (u16*)  alloc(512ULL*2048*2);
  float* pebsf = (float*)alloc(512ULL*2048*4);
  u16*   hedb  = (u16*)  alloc(512ULL*2048*2);
  u16*   h6b   = (u16*)  alloc(512ULL*2048*2);
  // big reusable regions
  u16* big0 = (u16*)alloc(512ULL*900*96*2);   // patches1, later out2 / out3
  u16* big1 = (u16*)alloc(512ULL*900*96*2);   // out1 (conv1 pre-pool NHWC)
  u16* patches1 = big0;
  u16* out1 = big1;
  u16* out2 = big0;
  u16* out3 = big0;

  auto ew = [&](long long total) { return (unsigned)((total + 255) / 256); };
  auto gemm = [&](const u16* A, int lda, const u16* Bt, int ldb,
                  float* of, int ldof, u16* ob, int ldob,
                  const float* bias, const float* mul, int ldmul,
                  int M, long long N, int K, int flags) {
    long long waves = (long long)(M / 16) * ((N + 63) / 64);
    unsigned blocks = (unsigned)((waves + 7) / 8);
    gemm_bf16_kernel<<<blocks, 256, 0, stream>>>(A, lda, Bt, ldb, of, ldof,
                                                 ob, ldob, bias, mul, ldmul,
                                                 M, N, K, flags);
  };

  // ---- weight packing (bf16, padded) ----
  pack_bf16_kernel<<<ew(96LL*96), 256, 0, stream>>>(w1r, conv1_w, 96, 75, 96, 96, 0, 0);
  pack_wr_kernel<<<ew(25LL*128*96), 256, 0, stream>>>(wr2, conv2_w, 128, 96);
  pack_wr_kernel<<<ew(25LL*256*128), 256, 0, stream>>>(wr3, conv3_w, 256, 128);
  pack_bf16_kernel<<<ew(2048LL*1024), 256, 0, stream>>>(fc1wb, fc1_w, 2048, 1024, 2048, 1024, 0, 0);
  pack_bf16_kernel<<<ew(2048LL*2048), 256, 0, stream>>>(fc2wb, fc2_w, 2048, 2048, 2048, 2048, 0, 0);
  pack_bf16_kernel<<<ew(64LL*2048), 256, 0, stream>>>(fc3wb, fc3_w, 10, 2048, 64, 2048, 0, 0);
  pack_bf16_kernel<<<ew(64LL*2048), 256, 0, stream>>>(wp3, fc3_w, 10, 2048, 64, 2048, 0, 1);
  pack_bf16_kernel<<<ew(2048LL*32), 256, 0, stream>>>(wp3t, fc3_w, 10, 2048, 2048, 32, 1, 1);
  pack_bf16_kernel<<<ew(2048LL*2048), 256, 0, stream>>>(wp2, fc2_w, 2048, 2048, 2048, 2048, 0, 1);
  pack_bf16_kernel<<<ew(2048LL*2048), 256, 0, stream>>>(wp2t, fc2_w, 2048, 2048, 2048, 2048, 1, 1);

  // ---- conv1: im2col + WMMA GEMM -> NHWC, relu+bias, then pool ----
  im2col1_kernel<<<ew(512LL*900*96), 256, 0, stream>>>(patches1, x);
  gemm(w1r, 96, patches1, 96, nullptr, 0, out1, 96,
       conv1_b, nullptr, 0, 96, 512LL*900, 96, EP_RELU | EP_BIAS_M | EP_STORE_T);
  maxpool3s2_kernel<<<ew(512LL*14*14*96), 256, 0, stream>>>(out1, h1, 512, 30, 30, 96, 14, 14);

  // ---- conv2 (implicit GEMM) + pool ----
  {
    long long waves = (128 / 16) * (512LL*14*14 / 64);
    conv5x5_wmma_kernel<<<(unsigned)((waves + 7) / 8), 256, 0, stream>>>(
        h1, wr2, conv2_b, out2, 512, 14, 14, 96, 128);
  }
  maxpool3s2_kernel<<<ew(512LL*6*6*128), 256, 0, stream>>>(out2, h2, 512, 14, 14, 128, 6, 6);

  // ---- conv3 (implicit GEMM) + pool + flatten ----
  {
    long long waves = (256 / 16) * (512LL*6*6 / 64);
    conv5x5_wmma_kernel<<<(unsigned)((waves + 7) / 8), 256, 0, stream>>>(
        h2, wr3, conv3_b, out3, 512, 6, 6, 128, 256);
  }
  maxpool3s2_kernel<<<ew(512LL*2*2*256), 256, 0, stream>>>(out3, h3, 512, 6, 6, 256, 2, 2);
  flatten_kernel<<<ew(512LL*1024), 256, 0, stream>>>(h3, h4);

  // ---- fc1: h5 = relu(h4 @ fc1_w^T + b) ----
  gemm(h4, 1024, fc1wb, 1024, h5f, 2048, h5b, 2048,
       fc1_b, nullptr, 0, 512, 2048, 1024, EP_RELU | EP_BIAS_N);
  // ---- h6_eb = relu(h5 @ fc2_w^T + b) ----
  gemm(h5b, 2048, fc2wb, 2048, h6ebf, 2048, h6ebb, 2048,
       fc2_b, nullptr, 0, 512, 2048, 2048, EP_RELU | EP_BIAS_N);
  // ---- EB through fc3: z = h6_eb @ Wp3^T ; s = one_hot/z ; p_h6 = h6_eb*(s@Wp3)
  gemm(h6ebb, 2048, wp3, 2048, zf, 16, nullptr, 0,
       nullptr, nullptr, 0, 512, 16, 2048, 0);
  s1_kernel<<<ew(512LL*32), 256, 0, stream>>>(zf, label, sb);
  gemm(sb, 32, wp3t, 32, ph6f, 2048, nullptr, 0,
       nullptr, h6ebf, 2048, 512, 2048, 32, 0);
  // ---- EB through fc2: z2 = h5 @ Wp2^T ; s2 = p_h6/z2 ; pebs = h5*(s2@Wp2)
  gemm(h5b, 2048, wp2, 2048, z2f, 2048, nullptr, 0,
       nullptr, nullptr, 0, 512, 2048, 2048, 0);
  s2_kernel<<<ew(512LL*2048), 256, 0, stream>>>(z2f, ph6f, s2b);
  gemm(s2b, 2048, wp2t, 2048, pebsf, 2048, nullptr, 0,
       nullptr, h5f, 2048, 512, 2048, 2048, 0);
  // ---- excitation dropout ----
  edrop_kernel<<<ew(512LL*2048), 256, 0, stream>>>(pebsf, noise, h5f, hedb);
  // ---- h6 = relu(h_ed @ fc2_w^T + b) ; out = h6 @ fc3_w^T + fc3_b ----
  gemm(hedb, 2048, fc2wb, 2048, nullptr, 0, h6b, 2048,
       fc2_b, nullptr, 0, 512, 2048, 2048, EP_RELU | EP_BIAS_N);
  gemm(h6b, 2048, fc3wb, 2048, out, 10, nullptr, 0,
       fc3_b, nullptr, 0, 512, 10, 2048, EP_BIAS_N);
}
